// EB_936302870591
// MI455X (gfx1250) — matrix operand order repaired and checked by reading, not compiled, not run on hardware
//
#include <hip/hip_runtime.h>
#include <hip/hip_bf16.h>

typedef _Float16 half_t;
typedef __attribute__((ext_vector_type(16))) _Float16 v16h;
typedef __attribute__((ext_vector_type(8)))  _Float16 v8h;
typedef __attribute__((ext_vector_type(8)))  float    v8f;
typedef __attribute__((ext_vector_type(4)))  float    v4f;

#define NH 32

__device__ __forceinline__ float psi_f(float t) {
    return copysignf(log1pf(fabsf(t)), t);
}

__device__ __forceinline__ float clip100_f(float t) {
    return __builtin_amdgcn_fmed3f(t, -100.0f, 100.0f); // v_med3_f32, 1 op
}

// pack 8 floats -> 8 halves (v_cvt_pk_f16_f32 x4), relu in f16 (v_pk_max_num_f16 x4)
__device__ __forceinline__ v8h pack_relu(v8f d) {
    v8h hv = __builtin_convertvector(d, v8h);
    v8h z  = (v8h)(half_t)0.0f;
    return __builtin_elementwise_max(hv, z);
}

__device__ __forceinline__ void wave_ds_fence() {
    asm volatile("s_wait_dscnt 0" ::: "memory");
}

// B fragment (32x16 f16, K x N): lane L -> column (base + L&15),
// halves K = (L<16 ? 0 : 16) .. +15, from a column-major (transposed) LDS copy.
__device__ __forceinline__ v16h load_bfrag(const half_t* __restrict__ Wt, int ncol, int lane) {
    int col = ncol + (lane & 15);
    int kb  = (lane < 16) ? 0 : 16;
    const half_t* p = Wt + col * NH + kb;
    union { v16h v; v8h h[2]; } u;
    u.h[0] = *(const v8h*)(p);
    u.h[1] = *(const v8h*)(p + 8);
    return u.v;
}

// A fragment (16x32 f16, M x K) from a row-major 32x32 LDS tile, row group t (0/1):
// lane L -> row t*16 + (L&15); kb = (L<16 ? 0 : 16); halves kb..kb+15.
__device__ __forceinline__ v16h load_afrag(const half_t* __restrict__ tile, int t, int lane) {
    int row = t * 16 + (lane & 15);
    int kb  = (lane < 16) ? 0 : 16;
    const half_t* p = tile + row * NH + kb;
    union { v16h v; v8h h[2]; } u;
    u.h[0] = *(const v8h*)(p);
    u.h[1] = *(const v8h*)(p + 8);
    return u.v;
}

__global__ void EB_zero_kernel(float* __restrict__ ws, int n) {
    int i = blockIdx.x * blockDim.x + threadIdx.x;
    if (i < n) ws[i] = 0.0f;
}

__global__ __launch_bounds__(256) void EB_edge_kernel(
    const float* __restrict__ x, const long long* __restrict__ ei,
    const float* __restrict__ W1, const float* __restrict__ bng,
    const float* __restrict__ bnb, const float* __restrict__ bnm,
    const float* __restrict__ bnv, const float* __restrict__ W2,
    const float* __restrict__ b2, const float* __restrict__ W3,
    const float* __restrict__ b3, const float* __restrict__ W4,
    const float* __restrict__ W5, const float* __restrict__ b5,
    float* __restrict__ m_out, float* __restrict__ sums,
    float* __restrict__ cnts, int n_edges)
{
    __shared__ __align__(16) half_t sW2t[NH * NH];   // transposed (col-major) f16
    __shared__ __align__(16) half_t sW3t[NH * NH];
    __shared__ __align__(16) half_t sW4h[NH];        // W4 as f16 for packed dot
    __shared__ __align__(16) half_t tA[8][NH * NH];  // per-wave 32x32 activation tile
    __shared__ __align__(16) half_t tB[8][NH * NH];  // per-wave 32x32 result tile
    __shared__ float sW1a[NH], sW1b[NH], sShift[NH]; // BN folded into layer 1
    __shared__ float sB2[NH], sB3[NH], sW5[NH];
    __shared__ float sB5;

    const int tid  = threadIdx.x;
    const int wave = tid >> 5;
    const int lane = tid & 31;
    const int laneR  = lane & 15;
    const int laneHi = lane >> 4;

    for (int idx = tid; idx < NH * NH; idx += 256) {
        int k = idx / NH, c = idx % NH;             // W[k][c] row-major in memory
        sW2t[c * NH + k] = (half_t)W2[idx];
        sW3t[c * NH + k] = (half_t)W3[idx];
    }
    if (tid < NH) {
        float sc = bng[tid] * rsqrtf(bnv[tid] + 1e-5f);
        sW1a[tid]   = W1[tid] * sc;                 // fold BN scale into W1
        sW1b[tid]   = W1[NH + tid] * sc;
        sShift[tid] = bnb[tid] - bnm[tid] * sc;
        sB2[tid]  = b2[tid];
        sB3[tid]  = b3[tid];
        sW4h[tid] = (half_t)W4[tid];
        sW5[tid]  = W5[tid];
    }
    if (tid == 0) sB5 = b5[0];
    __syncthreads();

    // Resident B fragments + biases + packed W4 (loop-invariant, kept in VGPRs)
    v16h B2f0 = load_bfrag(sW2t, 0, lane), B2f1 = load_bfrag(sW2t, 16, lane);
    v16h B3f0 = load_bfrag(sW3t, 0, lane), B3f1 = load_bfrag(sW3t, 16, lane);
    const float bias2[2] = { sB2[laneR], sB2[16 + laneR] };
    const float bias3[2] = { sB3[laneR], sB3[16 + laneR] };
    v8h w4p[4];
    #pragma unroll
    for (int q = 0; q < 4; ++q) w4p[q] = *(const v8h*)(sW4h + q * 8);

    half_t* myA = &tA[wave][0];
    half_t* myB = &tB[wave][0];

    const int numTiles = (n_edges + 255) >> 8;      // 256 edges per block-tile
    for (int tile = blockIdx.x; tile < numTiles; tile += gridDim.x) {
        const long long base = (long long)tile * 256 + wave * 32;
        const long long e    = base + lane;
        const bool valid     = e < (long long)n_edges;
        const long long ec   = valid ? e : 0;

        // prefetch next tile's indices into L2
        long long enext = e + (long long)gridDim.x * 256;
        if (enext < (long long)n_edges) {
            __builtin_prefetch(&ei[enext], 0, 1);
            __builtin_prefetch(&ei[(long long)n_edges + enext], 0, 1);
        }

        const long long vi = ei[ec];
        const long long vj = ei[(long long)n_edges + ec];
        const float xi0 = x[vi * 3 + 0], xi1 = x[vi * 3 + 1], xi2 = x[vi * 3 + 2];
        const float xj0 = x[vj * 3 + 0], xj1 = x[vj * 3 + 1], xj2 = x[vj * 3 + 2];
        const float dx0 = xi0 - xj0, dx1 = xi1 - xj1, dx2 = xi2 - xj2;
        const float nrm = psi_f(dx0 * dx0 + dx1 * dx1 + dx2 * dx2);
        const float dtp = psi_f(xi0 * xj0 + xi1 * xj1 + xi2 * xj2);

        // h = relu(nrm*W1a' + dtp*W1b' + shift) ; lane owns row `lane`
        #pragma unroll
        for (int q = 0; q < 4; ++q) {
            v8f hf;
            #pragma unroll
            for (int r = 0; r < 8; ++r) {
                int k = q * 8 + r;
                hf[r] = fmaf(nrm, sW1a[k], fmaf(dtp, sW1b[k], sShift[k]));
            }
            *(v8h*)(myA + lane * NH + q * 8) = pack_relu(hf);
        }
        wave_ds_fence();

        // GEMM1: h2 = relu(h @ W2 + b2)  -> myB (f16)
        #pragma unroll
        for (int t = 0; t < 2; ++t) {
            v16h A = load_afrag(myA, t, lane);
            #pragma unroll
            for (int n = 0; n < 2; ++n) {
                float bias = bias2[n];
                v8f c = {bias, bias, bias, bias, bias, bias, bias, bias};
                v8f d = __builtin_amdgcn_wmma_f32_16x16x32_f16(
                    false, A, false, (n == 0) ? B2f0 : B2f1, (short)0, c, false, false);
                v8h hv = pack_relu(d);
                int rowb = t * 16 + laneHi * 8;
                #pragma unroll
                for (int r = 0; r < 8; ++r)
                    myB[(rowb + r) * NH + n * 16 + laneR] = hv[r];
            }
        }
        wave_ds_fence();

        // gate: w = sigmoid(h2 . W5 + b5);  m = h2 * w  (store + feed GEMM2)
        float h2row[NH];
        float acc = sB5;
        #pragma unroll
        for (int q = 0; q < 4; ++q) {
            v8h pk = *(const v8h*)(myB + lane * NH + q * 8);
            #pragma unroll
            for (int r = 0; r < 8; ++r) {
                float v = (float)pk[r];
                h2row[q * 8 + r] = v;
                acc = fmaf(v, sW5[q * 8 + r], acc);
            }
        }
        const float wgt = 1.0f / (1.0f + __expf(-acc));
        float* mo = m_out + ec * NH;
        #pragma unroll
        for (int q = 0; q < 8; ++q) {
            v4f st;
            #pragma unroll
            for (int r = 0; r < 4; ++r) st[r] = h2row[q * 4 + r] * wgt;
            if (valid) __builtin_nontemporal_store(st, (v4f*)(mo + q * 4));
        }
        #pragma unroll
        for (int q = 0; q < 4; ++q) {
            v8f mf;
            #pragma unroll
            for (int r = 0; r < 8; ++r) mf[r] = h2row[q * 8 + r] * wgt;
            *(v8h*)(myA + lane * NH + q * 8) = __builtin_convertvector(mf, v8h);
        }
        wave_ds_fence();

        // GEMM2: h3 = relu(m @ W3 + b3) -> myB (f16)
        #pragma unroll
        for (int t = 0; t < 2; ++t) {
            v16h A = load_afrag(myA, t, lane);
            #pragma unroll
            for (int n = 0; n < 2; ++n) {
                float bias = bias3[n];
                v8f c = {bias, bias, bias, bias, bias, bias, bias, bias};
                v8f d = __builtin_amdgcn_wmma_f32_16x16x32_f16(
                    false, A, false, (n == 0) ? B3f0 : B3f1, (short)0, c, false, false);
                v8h hv = pack_relu(d);
                int rowb = t * 16 + laneHi * 8;
                #pragma unroll
                for (int r = 0; r < 8; ++r)
                    myB[(rowb + r) * NH + n * 16 + laneR] = hv[r];
            }
        }
        wave_ds_fence();

        // phix = h3 . W4 via packed f16 (v_pk_fma_f16), then horizontal reduce
        v8h pacc = (v8h)(half_t)0.0f;
        #pragma unroll
        for (int q = 0; q < 4; ++q) {
            v8h pk = *(const v8h*)(myB + lane * NH + q * 8);
            pacc += pk * w4p[q];
        }
        float phix = 0.0f;
        #pragma unroll
        for (int r = 0; r < 8; ++r) phix += (float)pacc[r];

        const float u0 = clip100_f(dx0 * phix);
        const float u1 = clip100_f(dx1 * phix);
        const float u2 = clip100_f(dx2 * phix);
        if (valid) {
            atomicAdd(&sums[vi * 3 + 0], u0);
            atomicAdd(&sums[vi * 3 + 1], u1);
            atomicAdd(&sums[vi * 3 + 2], u2);
            atomicAdd(&cnts[vi], 1.0f);
        }
    }
}

__global__ void EB_node_kernel(const float* __restrict__ x,
                               const float* __restrict__ sums,
                               const float* __restrict__ cnts,
                               float* __restrict__ xt, int n_nodes)
{
    int v = blockIdx.x * blockDim.x + threadIdx.x;
    if (v < n_nodes) {
        float c = fmaxf(cnts[v], 1.0f);
        float inv = 1.0f / c;
        xt[v * 3 + 0] = x[v * 3 + 0] + sums[v * 3 + 0] * inv;
        xt[v * 3 + 1] = x[v * 3 + 1] + sums[v * 3 + 1] * inv;
        xt[v * 3 + 2] = x[v * 3 + 2] + sums[v * 3 + 2] * inv;
    }
}

extern "C" void kernel_launch(void* const* d_in, const int* in_sizes, int n_in,
                              void* d_out, int out_size, void* d_ws, size_t ws_size,
                              hipStream_t stream) {
    const float*     x   = (const float*)d_in[0];
    const long long* ei  = (const long long*)d_in[1];
    const float*     W1  = (const float*)d_in[2];
    const float*     bng = (const float*)d_in[3];
    const float*     bnb = (const float*)d_in[4];
    const float*     bnm = (const float*)d_in[5];
    const float*     bnv = (const float*)d_in[6];
    const float*     W2  = (const float*)d_in[7];
    const float*     b2  = (const float*)d_in[8];
    const float*     W3  = (const float*)d_in[9];
    const float*     b3  = (const float*)d_in[10];
    const float*     W4  = (const float*)d_in[11];
    const float*     W5  = (const float*)d_in[12];
    const float*     b5  = (const float*)d_in[13];

    const int n_nodes = in_sizes[0] / 3;
    const int n_edges = in_sizes[1] / 2;

    float* xt    = (float*)d_out;                         // [N,3]
    float* m_out = (float*)d_out + (size_t)n_nodes * 3;   // [E,32]
    float* sums  = (float*)d_ws;                          // [N,3]
    float* cnts  = sums + (size_t)n_nodes * 3;            // [N]

    const int wsN = n_nodes * 4;
    EB_zero_kernel<<<(wsN + 255) / 256, 256, 0, stream>>>(sums, wsN);

    const int numTiles = (n_edges + 255) / 256;
    const int grid = numTiles < 4096 ? numTiles : 4096;
    EB_edge_kernel<<<grid, 256, 0, stream>>>(x, ei, W1, bng, bnb, bnm, bnv,
                                             W2, b2, W3, b3, W4, W5, b5,
                                             m_out, sums, cnts, n_edges);

    EB_node_kernel<<<(n_nodes + 255) / 256, 256, 0, stream>>>(x, sums, cnts, xt, n_nodes);
}
